// BlockConv_83700322664460
// MI455X (gfx1250) — matrix operand order repaired and checked by reading, not compiled, not run on hardware
//
#include <hip/hip_runtime.h>
#include <hip/hip_bf16.h>

// Problem constants (match reference)
#define GRID_N 256
#define BS     256
#define FEAT   256
#define BATCH  4
#define KDIAG  3

// WMMA vector types (wave32, CDNA5 layouts)
typedef __attribute__((ext_vector_type(16))) __bf16 v16bf;
typedef __attribute__((ext_vector_type(8)))  __bf16 v8bf;
typedef __attribute__((ext_vector_type(8)))  float  v8f;
typedef __attribute__((ext_vector_type(4)))  float  f32x4;

// B staging buffer: [64 feat rows][64 K cols + 16 pad] bf16, double buffered
#define BPAD 80
#define BBUF (64 * BPAD)

// blocks converted to bf16 once per launch (exact: entries are 0/1).
// Module-scope device buffer -> no workspace dependence, L2-resident (384 KB).
__device__ __align__(16) unsigned short g_abf[KDIAG * BS * BS];

__device__ __forceinline__ unsigned short bfbits(__bf16 h) {
    return __builtin_bit_cast(unsigned short, h);
}

// ---- prologue: blocks [3,256,256] fp32 -> bf16
__global__ __launch_bounds__(256)
void cvt_blocks_bf16(const float* __restrict__ blocks) {
    const int gid = blockIdx.x * 256 + threadIdx.x;      // 49152 threads, 4 elems each
    const f32x4 v = *(const f32x4*)(blocks + (size_t)gid * 4);
    unsigned int w0 = (unsigned int)bfbits((__bf16)v.x) | ((unsigned int)bfbits((__bf16)v.y) << 16);
    unsigned int w1 = (unsigned int)bfbits((__bf16)v.z) | ((unsigned int)bfbits((__bf16)v.w) << 16);
    unsigned int* dst = (unsigned int*)(g_abf + (size_t)gid * 4);
    dst[0] = w0;
    dst[1] = w1;
}

// One workgroup computes out[b, n*256 .. +255, f0 .. f0+63]:
//   out_block = sum_{d} blocks[d] @ x_block[n+d-1]   (zero at grid edges)
// A = blocks in bf16 (exact), B = x split hi+lo bf16, fp32 accumulate via
// v_wmma_f32_16x16x32_bf16.  B staging double-buffered through LDS.
__global__ __launch_bounds__(256)
void blockband_wmma(const float* __restrict__ x,
                    float* __restrict__ out) {
    __shared__ __align__(16) unsigned short Bhi[2 * BBUF];   // 20 KB
    __shared__ __align__(16) unsigned short Blo[2 * BBUF];   // 20 KB

    const int n    = blockIdx.x;
    const int b    = blockIdx.y;
    const int f0   = blockIdx.z * 64;
    const int tid  = threadIdx.x;
    const int lane = tid & 31;
    const int wave = tid >> 5;              // wave owns output rows [32w, 32w+32)

    // valid diagonals form a contiguous range [dlo, dhi] (uniform, branch-free)
    const int dlo = (n == 0) ? 1 : 0;
    const int dhi = (n == GRID_N - 1) ? 1 : 2;
    const int C   = (dhi - dlo + 1) * 4;    // 64-wide K chunks to process

    const v8f vzero = {0.f, 0.f, 0.f, 0.f, 0.f, 0.f, 0.f, 0.f};
    v8f acc[2][4];
    #pragma unroll
    for (int mi = 0; mi < 2; ++mi)
        #pragma unroll
        for (int ni = 0; ni < 4; ++ni) acc[mi][ni] = vzero;

    // ---- B staging: thread owns K-row pair (j0, j0+1) x 8 feature cols
    const int j0 = (tid & 31) << 1;
    const int fb = (tid >> 5) * 8;
    f32x4 rg[4];

    auto load_regs = [&](int c) {
        const int d  = dlo + (c >> 2);
        const int nb = n + d - 1;
        const int kk = (c & 3) << 6;
        const float* s = x + (((size_t)b * (GRID_N * BS))
                              + (size_t)nb * BS + kk + j0) * FEAT + f0 + fb;
        rg[0] = *(const f32x4*)(s);
        rg[1] = *(const f32x4*)(s + 4);
        rg[2] = *(const f32x4*)(s + FEAT);
        rg[3] = *(const f32x4*)(s + FEAT + 4);
        if ((c & 3) != 3)                                  // CDNA5 global_prefetch_b8
            __builtin_prefetch(s + (size_t)64 * FEAT, 0, 1);
    };

    auto cvt_store = [&](int p) {
        unsigned short* bh = Bhi + p * BBUF;
        unsigned short* bl = Blo + p * BBUF;
        float a0[8], a1[8];
        #pragma unroll
        for (int q = 0; q < 4; ++q) {
            a0[q] = rg[0][q]; a0[4 + q] = rg[1][q];        // K row j0
            a1[q] = rg[2][q]; a1[4 + q] = rg[3][q];        // K row j0+1
        }
        #pragma unroll
        for (int c2 = 0; c2 < 8; ++c2) {
            const __bf16 h0 = (__bf16)a0[c2];
            const __bf16 h1 = (__bf16)a1[c2];
            const __bf16 l0 = (__bf16)(a0[c2] - (float)h0);
            const __bf16 l1 = (__bf16)(a1[c2] - (float)h1);
            const int off = (fb + c2) * BPAD + j0;         // j0 even -> 4B aligned
            *(unsigned int*)(bh + off) = (unsigned int)bfbits(h0) | ((unsigned int)bfbits(h1) << 16);
            *(unsigned int*)(bl + off) = (unsigned int)bfbits(l0) | ((unsigned int)bfbits(l1) << 16);
        }
    };

    auto compute = [&](int c, int p) {
        const int d  = dlo + (c >> 2);
        const int kk = (c & 3) << 6;
        const unsigned short* bhB = Bhi + p * BBUF;
        const unsigned short* blB = Blo + p * BBUF;
        #pragma unroll
        for (int ks = 0; ks < 64; ks += 32) {
            v16bf afr[2], bh[4], bl[4];

            // A fragments straight from L2 bf16 copy (per-lane layout:
            // K = {base, base+16}+0..7, base = kk+ks + (lane<16 ? 0 : 8))
            #pragma unroll
            for (int mi = 0; mi < 2; ++mi) {
                const int row = wave * 32 + mi * 16 + (lane & 15);
                const int kg  = kk + ks + ((lane < 16) ? 0 : 8);
                const unsigned short* pa = g_abf + ((size_t)d * BS + row) * BS + kg;
                v8bf a0v = *(const v8bf*)(pa);
                v8bf a1v = *(const v8bf*)(pa + 16);
                #pragma unroll
                for (int e = 0; e < 8; ++e) {
                    afr[mi][e]     = a0v[e];
                    afr[mi][e + 8] = a1v[e];
                }
            }

            // B fragments: lane holds col (l&15); K = {base, base+16}+0..7, base = (l<16?0:8)
            #pragma unroll
            for (int ni = 0; ni < 4; ++ni) {
                const int f    = ni * 16 + (lane & 15);
                const int koff = ks + ((lane < 16) ? 0 : 8);
                const unsigned short* ph = bhB + f * BPAD + koff;
                v8bf h0 = *(const v8bf*)(ph);
                v8bf h1 = *(const v8bf*)(ph + 16);
                const unsigned short* pl = blB + f * BPAD + koff;
                v8bf l0 = *(const v8bf*)(pl);
                v8bf l1 = *(const v8bf*)(pl + 16);
                #pragma unroll
                for (int e = 0; e < 8; ++e) {
                    bh[ni][e]     = h0[e];
                    bh[ni][e + 8] = h1[e];
                    bl[ni][e]     = l0[e];
                    bl[ni][e + 8] = l1[e];
                }
            }

            // 16 WMMAs: acc += A*Bhi + A*Blo (fp32 accumulate)
            #pragma unroll
            for (int mi = 0; mi < 2; ++mi)
                #pragma unroll
                for (int ni = 0; ni < 4; ++ni) {
                    acc[mi][ni] = __builtin_amdgcn_wmma_f32_16x16x32_bf16(
                        false, afr[mi], false, bh[ni],
                        (short)0, acc[mi][ni], false, false);
                    acc[mi][ni] = __builtin_amdgcn_wmma_f32_16x16x32_bf16(
                        false, afr[mi], false, bl[ni],
                        (short)0, acc[mi][ni], false, false);
                }
        }
    };

    // ---- software-pipelined main loop: stage c+1 while computing c
    load_regs(0);
    cvt_store(0);
    __syncthreads();
    for (int c = 0; c < C; ++c) {
        if (c + 1 < C) load_regs(c + 1);        // global loads for next chunk in flight
        compute(c, c & 1);                      // WMMA burst on current buffer
        if (c + 1 < C) cvt_store((c + 1) & 1);  // fill opposite buffer
        __syncthreads();
    }

    // ---- store: C/D layout — VGPR r => row r + (l<16?0:8); col = l&15
    #pragma unroll
    for (int mi = 0; mi < 2; ++mi) {
        const int row0 = n * BS + wave * 32 + mi * 16 + ((lane < 16) ? 0 : 8);
        #pragma unroll
        for (int ni = 0; ni < 4; ++ni) {
            const int col = f0 + ni * 16 + (lane & 15);
            float* dst = out + ((size_t)b * (GRID_N * BS) + row0) * FEAT + col;
            #pragma unroll
            for (int r = 0; r < 8; ++r)
                dst[(size_t)r * FEAT] = acc[mi][ni][r];
        }
    }
}

extern "C" void kernel_launch(void* const* d_in, const int* in_sizes, int n_in,
                              void* d_out, int out_size, void* d_ws, size_t ws_size,
                              hipStream_t stream) {
    const float* x      = (const float*)d_in[0];   // [4, 65536, 256] fp32
    const float* blocks = (const float*)d_in[1];   // [3, 256, 256]  fp32
    float* out          = (float*)d_out;           // [4, 65536, 256] fp32
    (void)d_ws; (void)ws_size;

    // prologue: refresh bf16 copy of blocks every call (deterministic)
    cvt_blocks_bf16<<<dim3((KDIAG * BS * BS / 4) / 256), dim3(256), 0, stream>>>(blocks);

    dim3 grid(GRID_N, BATCH, FEAT / 64);           // (n, b, feature-chunk)
    dim3 block(256);                               // 8 wave32s
    blockband_wmma<<<grid, block, 0, stream>>>(x, out);
}
